// DecoderLayer_29051158790750
// MI455X (gfx1250) — compile-verified, matrix operand order checked by reading, not run on hardware
//
#include <hip/hip_runtime.h>

// ---------------- problem constants ----------------
#define HID   1024
#define HEADS 16
#define INTER 4096
#define HD    64          // head dim
#define SEQ   2048
#define BATCH 2
#define TOK   (BATCH * SEQ)   // 4096 tokens
#define EPSV  1e-6f
#define SCALEV 0.125f         // 64^-0.5

#define TPB   256
#define WAVES (TPB / 32)

typedef __attribute__((ext_vector_type(4)))  unsigned int v4u;
typedef __attribute__((ext_vector_type(16))) __bf16       v16bf;
typedef __attribute__((ext_vector_type(8)))  float        v8f;

union Frag {
    v16bf v;
    struct { v4u lo; v4u hi; } q;
    unsigned int d[8];
};

// float -> bf16 bits, round-to-nearest-even
__device__ __forceinline__ unsigned short f2bf(float x) {
    unsigned int u = __float_as_uint(x);
    return (unsigned short)((u + 0x7FFFu + ((u >> 16) & 1u)) >> 16);
}
__device__ __forceinline__ float bf2f(unsigned short b) {
    return __uint_as_float(((unsigned int)b) << 16);
}

// ---- WMMA bf16 16x16x32 fragment layout (ISA 7.12.2, wave32) ----
// A: lane L holds row m = L&15.  VGPR v holds K pair:
//   k = (v>=4 ? 16 : 0) + (L>>4)*8 + (v&3)*2  (+0,+1 in lo/hi halves)
// B assumed symmetric: lane L holds column n = L&15 with identical K layout.
__device__ __forceinline__ v16bf load_frag_bf16(const unsigned short* __restrict__ row,
                                                int k0, int half) {
    Frag f;
    const unsigned short* p = row + k0 + half * 8;   // 16B aligned
    f.q.lo = *(const v4u*)(p);
    f.q.hi = *(const v4u*)(p + 16);
    return f.v;
}

// Same fragment but sourced from fp32 memory, converting to bf16 on load.
__device__ __forceinline__ v16bf load_frag_f32(const float* __restrict__ row,
                                               int k0, int half) {
    Frag f;
    const float* p = row + k0 + half * 8;
#pragma unroll
    for (int g = 0; g < 2; ++g) {
#pragma unroll
        for (int j = 0; j < 4; ++j) {
            float a = p[g * 16 + j * 2];
            float b = p[g * 16 + j * 2 + 1];
            f.d[g * 4 + j] = (unsigned int)f2bf(a) | ((unsigned int)f2bf(b) << 16);
        }
    }
    return f.v;
}

// 64x64 register-blocked WMMA core: 16 accumulators, 1 b128 load per WMMA.
__device__ __forceinline__ void wmma_tile64(const unsigned short* const arow[4],
                                            const unsigned short* const brow[4],
                                            int K, int half, v8f c[4][4]) {
    for (int k0 = 0; k0 < K; k0 += 32) {
        v16bf a[4], b[4];
#pragma unroll
        for (int i = 0; i < 4; ++i) a[i] = load_frag_bf16(arow[i], k0, half);
#pragma unroll
        for (int j = 0; j < 4; ++j) b[j] = load_frag_bf16(brow[j], k0, half);
#pragma unroll
        for (int i = 0; i < 4; ++i)
#pragma unroll
            for (int j = 0; j < 4; ++j)
                c[i][j] = __builtin_amdgcn_wmma_f32_16x16x32_bf16(
                    false, a[i], false, b[j], (short)0, c[i][j], false, false);
    }
}

// ---------------- weight transpose + bf16 convert:  Wt[n][k] = bf16(W[k][n]) ----------------
__global__ __launch_bounds__(TPB) void transpose_w(const float* __restrict__ W,
                                                   unsigned short* __restrict__ Wt,
                                                   int K, int N) {
    size_t idx = (size_t)blockIdx.x * blockDim.x + threadIdx.x;
    if (idx >= (size_t)K * N) return;
    int k = (int)(idx / N);
    int n = (int)(idx % N);
    Wt[(size_t)n * K + k] = f2bf(W[idx]);
}

// ---------------- RMSNorm: one block per token row of HID ----------------
__global__ __launch_bounds__(TPB) void rmsnorm_k(const float* __restrict__ X,
                                                 const float* __restrict__ g,
                                                 unsigned short* __restrict__ Out) {
    __shared__ float red[TPB];
    size_t row = blockIdx.x;
    const float* x = X + row * HID;
    int t = threadIdx.x;
    float ss = 0.f;
    for (int i = t; i < HID; i += TPB) { float v = x[i]; ss += v * v; }
    red[t] = ss; __syncthreads();
    for (int s = TPB / 2; s > 0; s >>= 1) {
        if (t < s) red[t] += red[t + s];
        __syncthreads();
    }
    float r = rsqrtf(red[0] * (1.0f / HID) + EPSV);
    for (int i = t; i < HID; i += TPB)
        Out[row * HID + i] = f2bf(x[i] * r * (1.0f + g[i]));
}

// ---------------- blocked WMMA GEMM: C(bf16) = A(MxK,bf16) @ Bt(NxK,bf16)^T + bias ----------------
__global__ __launch_bounds__(TPB) void gemm_bf16(const unsigned short* __restrict__ A,
                                                 const unsigned short* __restrict__ Bt,
                                                 const float* __restrict__ bias,
                                                 unsigned short* __restrict__ C,
                                                 int M, int N, int K) {
    int wave = threadIdx.x >> 5, lane = threadIdx.x & 31;
    int half = lane >> 4, li = lane & 15;
    int ntiles = N >> 6;
    int tile = blockIdx.x * WAVES + wave;
    if (tile >= (M >> 6) * ntiles) return;
    int mt = tile / ntiles, nt = tile % ntiles;

    const unsigned short* arow[4];
    const unsigned short* brow[4];
#pragma unroll
    for (int i = 0; i < 4; ++i) arow[i] = A + (size_t)(mt * 64 + i * 16 + li) * K;
#pragma unroll
    for (int j = 0; j < 4; ++j) brow[j] = Bt + (size_t)(nt * 64 + j * 16 + li) * K;

    v8f c[4][4] = {};
    wmma_tile64(arow, brow, K, half, c);

#pragma unroll
    for (int j = 0; j < 4; ++j) {
        int n = nt * 64 + j * 16 + li;
        float bv = bias[n];
#pragma unroll
        for (int i = 0; i < 4; ++i)
#pragma unroll
            for (int r = 0; r < 8; ++r) {
                int m = mt * 64 + i * 16 + half * 8 + r;
                C[(size_t)m * N + n] = f2bf(c[i][j][r] + bv);
            }
    }
}

// ---------------- V projection with head-transposed store: VT[b][h][d][tok] ----------------
// N blocked by 64 == HD, so n-block nt maps exactly to head nt.
__global__ __launch_bounds__(TPB) void gemm_vt(const unsigned short* __restrict__ A,
                                               const unsigned short* __restrict__ Bt,
                                               const float* __restrict__ bias,
                                               unsigned short* __restrict__ VT) {
    int wave = threadIdx.x >> 5, lane = threadIdx.x & 31;
    int half = lane >> 4, li = lane & 15;
    const int ntiles = HID >> 6;          // 16 heads
    int tile = blockIdx.x * WAVES + wave;
    if (tile >= (TOK >> 6) * ntiles) return;
    int mt = tile / ntiles, nt = tile % ntiles;   // nt == head

    const unsigned short* arow[4];
    const unsigned short* brow[4];
#pragma unroll
    for (int i = 0; i < 4; ++i) arow[i] = A + (size_t)(mt * 64 + i * 16 + li) * HID;
#pragma unroll
    for (int j = 0; j < 4; ++j) brow[j] = Bt + (size_t)(nt * 64 + j * 16 + li) * HID;

    v8f c[4][4] = {};
    wmma_tile64(arow, brow, HID, half, c);

#pragma unroll
    for (int j = 0; j < 4; ++j) {
        int n = nt * 64 + j * 16 + li;
        int d = j * 16 + li;
        float bv = bias[n];
#pragma unroll
        for (int i = 0; i < 4; ++i)
#pragma unroll
            for (int r = 0; r < 8; ++r) {
                int m = mt * 64 + i * 16 + half * 8 + r;   // global token index
                int b = m >> 11;
                int tk = m & (SEQ - 1);
                VT[(((size_t)b * HEADS + nt) * HD + d) * SEQ + tk] = f2bf(c[i][j][r] + bv);
            }
    }
}

// ---------------- attention scores: P = scale * Q K^T, masked, fp32 into d_out ----------------
__global__ __launch_bounds__(TPB) void attn_scores(const unsigned short* __restrict__ Q,
                                                   const unsigned short* __restrict__ Kb,
                                                   const int* __restrict__ mask,
                                                   float* __restrict__ P) {
    int wave = threadIdx.x >> 5, lane = threadIdx.x & 31;
    int half = lane >> 4, li = lane & 15;
    const int tilesN = SEQ / 64;                    // 32
    const int tilesPerBH = tilesN * tilesN;         // 1024
    int tile = blockIdx.x * WAVES + wave;
    if (tile >= BATCH * HEADS * tilesPerBH) return;
    int bh = tile / tilesPerBH;
    int t  = tile % tilesPerBH;
    int mt = t / tilesN, nt = t % tilesN;
    int b = bh / HEADS, h = bh % HEADS;

    const unsigned short* arow[4];
    const unsigned short* brow[4];
#pragma unroll
    for (int i = 0; i < 4; ++i)
        arow[i] = Q + ((size_t)(b * SEQ + mt * 64 + i * 16 + li)) * HID + h * HD;
#pragma unroll
    for (int j = 0; j < 4; ++j)
        brow[j] = Kb + ((size_t)(b * SEQ + nt * 64 + j * 16 + li)) * HID + h * HD;

    v8f c[4][4] = {};
    wmma_tile64(arow, brow, HD, half, c);          // K = 64, 2 steps

    float* prow = P + (size_t)bh * SEQ * SEQ;
    const int* mrow = mask + (size_t)b * SEQ * SEQ;
#pragma unroll
    for (int i = 0; i < 4; ++i)
#pragma unroll
        for (int j = 0; j < 4; ++j) {
            int n = nt * 64 + j * 16 + li;
#pragma unroll
            for (int r = 0; r < 8; ++r) {
                int m = mt * 64 + i * 16 + half * 8 + r;
                float s = c[i][j][r] * SCALEV;
                if (mrow[(size_t)m * SEQ + n] == 0) s = -10000.0f;
                prow[(size_t)m * SEQ + n] = s;
            }
        }
}

// ---------------- row softmax in place (one block per row of SEQ) ----------------
__global__ __launch_bounds__(TPB) void softmax_rows(float* __restrict__ P) {
    __shared__ float red[TPB];
    float* p = P + (size_t)blockIdx.x * SEQ;
    int t = threadIdx.x;
    float mx = -3.0e38f;
    for (int i = t; i < SEQ; i += TPB) mx = fmaxf(mx, p[i]);
    red[t] = mx; __syncthreads();
    for (int s = TPB / 2; s > 0; s >>= 1) {
        if (t < s) red[t] = fmaxf(red[t], red[t + s]);
        __syncthreads();
    }
    mx = red[0]; __syncthreads();
    float sum = 0.f;
    for (int i = t; i < SEQ; i += TPB) {
        float e = __expf(p[i] - mx);
        p[i] = e; sum += e;
    }
    red[t] = sum; __syncthreads();
    for (int s = TPB / 2; s > 0; s >>= 1) {
        if (t < s) red[t] += red[t + s];
        __syncthreads();
    }
    float inv = 1.0f / red[0];
    for (int i = t; i < SEQ; i += TPB) p[i] *= inv;
}

// ---------------- attn = P @ V : fp32 probs (cvt on load) x bf16 VT -> bf16 token-major ----------------
// Per (b,h): M=SEQ, N=HD=64 -> one 4-wide n-block per head.
__global__ __launch_bounds__(TPB) void attn_pv(const float* __restrict__ P,
                                               const unsigned short* __restrict__ VT,
                                               unsigned short* __restrict__ Out) {
    int wave = threadIdx.x >> 5, lane = threadIdx.x & 31;
    int half = lane >> 4, li = lane & 15;
    const int tilesPerBH = SEQ / 64;                // 32 (n covered by one block)
    int tile = blockIdx.x * WAVES + wave;
    if (tile >= BATCH * HEADS * tilesPerBH) return;
    int bh = tile / tilesPerBH;
    int mt = tile % tilesPerBH;
    int b = bh / HEADS, h = bh % HEADS;

    const float* arow[4];
    const unsigned short* brow[4];
#pragma unroll
    for (int i = 0; i < 4; ++i)
        arow[i] = P + ((size_t)bh * SEQ + mt * 64 + i * 16 + li) * SEQ;
#pragma unroll
    for (int j = 0; j < 4; ++j)
        brow[j] = VT + ((size_t)bh * HD + j * 16 + li) * SEQ;

    v8f c[4][4] = {};
    for (int k0 = 0; k0 < SEQ; k0 += 32) {
        v16bf a[4], v[4];
#pragma unroll
        for (int i = 0; i < 4; ++i) a[i] = load_frag_f32(arow[i], k0, half);
#pragma unroll
        for (int j = 0; j < 4; ++j) v[j] = load_frag_bf16(brow[j], k0, half);
#pragma unroll
        for (int i = 0; i < 4; ++i)
#pragma unroll
            for (int j = 0; j < 4; ++j)
                c[i][j] = __builtin_amdgcn_wmma_f32_16x16x32_bf16(
                    false, a[i], false, v[j], (short)0, c[i][j], false, false);
    }

#pragma unroll
    for (int j = 0; j < 4; ++j) {
        int d = j * 16 + li;
#pragma unroll
        for (int i = 0; i < 4; ++i)
#pragma unroll
            for (int r = 0; r < 8; ++r) {
                int m = mt * 64 + i * 16 + half * 8 + r;
                Out[((size_t)(b * SEQ) + m) * HID + h * HD + d] = f2bf(c[i][j][r]);
            }
    }
}

// ---------------- blocked WMMA GEMM, fp32 output + fused residual ----------------
__global__ __launch_bounds__(TPB) void gemm_f32_res(const unsigned short* __restrict__ A,
                                                    const unsigned short* __restrict__ Bt,
                                                    const float* __restrict__ bias,
                                                    const float* __restrict__ res,
                                                    float* __restrict__ C,
                                                    int M, int N, int K) {
    int wave = threadIdx.x >> 5, lane = threadIdx.x & 31;
    int half = lane >> 4, li = lane & 15;
    int ntiles = N >> 6;
    int tile = blockIdx.x * WAVES + wave;
    if (tile >= (M >> 6) * ntiles) return;
    int mt = tile / ntiles, nt = tile % ntiles;

    const unsigned short* arow[4];
    const unsigned short* brow[4];
#pragma unroll
    for (int i = 0; i < 4; ++i) arow[i] = A + (size_t)(mt * 64 + i * 16 + li) * K;
#pragma unroll
    for (int j = 0; j < 4; ++j) brow[j] = Bt + (size_t)(nt * 64 + j * 16 + li) * K;

    v8f c[4][4] = {};
    wmma_tile64(arow, brow, K, half, c);

#pragma unroll
    for (int j = 0; j < 4; ++j) {
        int n = nt * 64 + j * 16 + li;
        float bv = bias[n];
#pragma unroll
        for (int i = 0; i < 4; ++i)
#pragma unroll
            for (int r = 0; r < 8; ++r) {
                int m = mt * 64 + i * 16 + half * 8 + r;
                size_t idx = (size_t)m * N + n;
                C[idx] = c[i][j][r] + bv + res[idx];
            }
    }
}

// ---------------- gated GELU (tanh approx): G = gelu(G) * U, bf16 in/out ----------------
__global__ __launch_bounds__(TPB) void gelu_mul(unsigned short* __restrict__ G,
                                                const unsigned short* __restrict__ U,
                                                size_t n) {
    size_t i = (size_t)blockIdx.x * blockDim.x + threadIdx.x;
    if (i >= n) return;
    float g = bf2f(G[i]);
    float u = bf2f(U[i]);
    float inner = 0.7978845608028654f * (g + 0.044715f * g * g * g);
    float y = 0.5f * g * (1.0f + tanhf(inner));
    G[i] = f2bf(y * u);
}

// ---------------- host launcher ----------------
extern "C" void kernel_launch(void* const* d_in, const int* in_sizes, int n_in,
                              void* d_out, int out_size, void* d_ws, size_t ws_size,
                              hipStream_t stream) {
    const float* x    = (const float*)d_in[0];
    const int*   mask = (const int*)d_in[1];
    const float* g1   = (const float*)d_in[2];
    const float* g2   = (const float*)d_in[3];
    const float* wq   = (const float*)d_in[4];
    const float* bq   = (const float*)d_in[5];
    const float* wk   = (const float*)d_in[6];
    const float* bk   = (const float*)d_in[7];
    const float* wv   = (const float*)d_in[8];
    const float* bv   = (const float*)d_in[9];
    const float* wo   = (const float*)d_in[10];
    const float* bo   = (const float*)d_in[11];
    const float* wg   = (const float*)d_in[12];
    const float* bg   = (const float*)d_in[13];
    const float* wu   = (const float*)d_in[14];
    const float* bu   = (const float*)d_in[15];
    const float* wd   = (const float*)d_in[16];
    const float* bd   = (const float*)d_in[17];

    float* out_x = (float*)d_out;                          // (2,2048,1024) fp32
    float* probs = out_x + (size_t)TOK * HID;              // (2,16,2048,2048) fp32

    char* p = (char*)d_ws;
    auto take = [&](size_t bytes) { char* q = p; p += (bytes + 255) & ~(size_t)255; return q; };
    unsigned short* wqT = (unsigned short*)take((size_t)HID * HID * 2);
    unsigned short* wkT = (unsigned short*)take((size_t)HID * HID * 2);
    unsigned short* wvT = (unsigned short*)take((size_t)HID * HID * 2);
    unsigned short* woT = (unsigned short*)take((size_t)HID * HID * 2);
    unsigned short* wgT = (unsigned short*)take((size_t)HID * INTER * 2);
    unsigned short* wuT = (unsigned short*)take((size_t)HID * INTER * 2);
    unsigned short* wdT = (unsigned short*)take((size_t)INTER * HID * 2);
    unsigned short* xn   = (unsigned short*)take((size_t)TOK * HID * 2);
    unsigned short* qb   = (unsigned short*)take((size_t)TOK * HID * 2);
    unsigned short* kb   = (unsigned short*)take((size_t)TOK * HID * 2);
    unsigned short* vt   = (unsigned short*)take((size_t)TOK * HID * 2);
    unsigned short* attn = (unsigned short*)take((size_t)TOK * HID * 2);
    float*          x1   = (float*)take((size_t)TOK * HID * 4);
    unsigned short* xn2  = (unsigned short*)take((size_t)TOK * HID * 2);
    unsigned short* gbuf = (unsigned short*)take((size_t)TOK * INTER * 2);
    unsigned short* ubuf = (unsigned short*)take((size_t)TOK * INTER * 2);

    auto cdiv = [](size_t a, size_t b) { return (int)((a + b - 1) / b); };

    // 1) weight transpose + bf16 convert
    transpose_w<<<cdiv((size_t)HID * HID, TPB), TPB, 0, stream>>>(wq, wqT, HID, HID);
    transpose_w<<<cdiv((size_t)HID * HID, TPB), TPB, 0, stream>>>(wk, wkT, HID, HID);
    transpose_w<<<cdiv((size_t)HID * HID, TPB), TPB, 0, stream>>>(wv, wvT, HID, HID);
    transpose_w<<<cdiv((size_t)HID * HID, TPB), TPB, 0, stream>>>(wo, woT, HID, HID);
    transpose_w<<<cdiv((size_t)HID * INTER, TPB), TPB, 0, stream>>>(wg, wgT, HID, INTER);
    transpose_w<<<cdiv((size_t)HID * INTER, TPB), TPB, 0, stream>>>(wu, wuT, HID, INTER);
    transpose_w<<<cdiv((size_t)INTER * HID, TPB), TPB, 0, stream>>>(wd, wdT, INTER, HID);

    // 2) pre-attention RMSNorm
    rmsnorm_k<<<TOK, TPB, 0, stream>>>(x, g1, xn);

    // 3) Q, K projections (token-major bf16), V projection (head-transposed)
    int projBlocks = ((TOK / 64) * (HID / 64)) / WAVES;      // 128
    gemm_bf16<<<projBlocks, TPB, 0, stream>>>(xn, wqT, bq, qb, TOK, HID, HID);
    gemm_bf16<<<projBlocks, TPB, 0, stream>>>(xn, wkT, bk, kb, TOK, HID, HID);
    gemm_vt  <<<projBlocks, TPB, 0, stream>>>(xn, wvT, bv, vt);

    // 4) scores -> probs region of d_out (pre-softmax), then 5) softmax in place
    int scoreBlocks = (BATCH * HEADS * (SEQ / 64) * (SEQ / 64)) / WAVES;   // 4096
    attn_scores<<<scoreBlocks, TPB, 0, stream>>>(qb, kb, mask, probs);
    softmax_rows<<<BATCH * HEADS * SEQ, TPB, 0, stream>>>(probs);

    // 6) attn = probs @ V
    int pvBlocks = (BATCH * HEADS * (SEQ / 64)) / WAVES;                   // 128
    attn_pv<<<pvBlocks, TPB, 0, stream>>>(probs, vt, attn);

    // 7) output projection + residual -> x1 (fp32)
    gemm_f32_res<<<projBlocks, TPB, 0, stream>>>(attn, woT, bo, x, x1, TOK, HID, HID);

    // 8) pre-MLP RMSNorm
    rmsnorm_k<<<TOK, TPB, 0, stream>>>(x1, g2, xn2);

    // 9) gate & up projections
    int mlpBlocks = ((TOK / 64) * (INTER / 64)) / WAVES;     // 512
    gemm_bf16<<<mlpBlocks, TPB, 0, stream>>>(xn2, wgT, bg, gbuf, TOK, INTER, HID);
    gemm_bf16<<<mlpBlocks, TPB, 0, stream>>>(xn2, wuT, bu, ubuf, TOK, INTER, HID);

    // 10) gated GELU (in place over gbuf)
    gelu_mul<<<cdiv((size_t)TOK * INTER, TPB), TPB, 0, stream>>>(gbuf, ubuf, (size_t)TOK * INTER);

    // 11) down projection + residual -> final x output
    gemm_f32_res<<<projBlocks, TPB, 0, stream>>>(gbuf, wdT, bd, x1, out_x, TOK, HID, INTER);
}
